// Trans_Attn_28767690948650
// MI455X (gfx1250) — compile-verified, matrix operand order checked by reading, not run on hardware
//
#include <hip/hip_runtime.h>
#include <hip/hip_bf16.h>

typedef __attribute__((ext_vector_type(16))) _Float16     v16h;
typedef __attribute__((ext_vector_type(8)))  float        v8f;
typedef __attribute__((ext_vector_type(4)))  unsigned int v4u;
typedef __attribute__((ext_vector_type(8)))  int          v8i;
typedef __attribute__((ext_vector_type(4)))  int          v4i;

#define BDIM 2
#define CDIM 256
#define CQK  32
#define NPIX 4096
#define EPSV 1e-5f

#if defined(__has_include)
#if __has_include(<hip/amd_detail/amd_gfx1250_TDM.h>)
#define TDM_6ARG 1
#endif
#endif

// K-index map for 16-bit A/B WMMA fragments (16 halfs per lane, wave32):
// e=0..7  -> K = half*8 + e          (contiguous 16B run)
// e=8..15 -> K = 16 + half*8 + (e-8) (contiguous 16B run)
__device__ __forceinline__ int kmap(int half, int e) {
  return ((e >> 3) << 4) | (half << 3) | (((e >> 1) & 3) << 1) | (e & 1);
}

// LDS byte offset of a __shared__ object (generic-pointer low 32 bits).
__device__ __forceinline__ unsigned lds_off(const void* p) {
  return (unsigned)(unsigned long long)(size_t)p;
}

// ---------------------------------------------------------------------------
// Tensor Data Mover: DMA a 2-D tile [tile_h rows x tile_w elems], row stride
// stride_elems, element size = 2^dsz bytes (1 -> f16), global -> LDS.
// Descriptor packing per CDNA5 ISA sec 8.3/8.4. Wave-uniform args only.
// ---------------------------------------------------------------------------
__device__ __forceinline__ void tdm_load_2d(unsigned lds_byte_off, const void* gptr,
                                            unsigned tile_w, unsigned tile_h,
                                            unsigned long long stride_elems,
                                            unsigned dsz) {
  unsigned long long ga = (unsigned long long)(size_t)gptr;
  const unsigned td0 = (stride_elems > tile_w) ? (unsigned)stride_elems : tile_w;
  const unsigned td1 = tile_h;

  v4u g0;
  g0.x = 1u;                                                 // count=1, user
  g0.y = lds_byte_off;                                       // lds_addr
  g0.z = (unsigned)(ga & 0xffffffffu);                       // global_addr lo
  g0.w = (unsigned)((ga >> 32) & 0x01ffffffu) | (2u << 30);  // hi + type=2

  v8i g1;
  g1[0] = (int)(dsz << 16);                                  // data_size
  g1[1] = (int)((td0 & 0xffffu) << 16);                      // tensor_dim0 lo16
  g1[2] = (int)(((td0 >> 16) & 0xffffu) | ((td1 & 0xffffu) << 16));
  g1[3] = (int)(((td1 >> 16) & 0xffffu) | (tile_w << 16));   // tile_dim0
  g1[4] = (int)tile_h;                                       // tile_dim1
  g1[5] = (int)(stride_elems & 0xffffffffu);                 // dim0 stride lo
  g1[6] = (int)((stride_elems >> 32) & 0xffffu);             // dim0 stride hi
  g1[7] = 0;

  v4i g2 = {0, 0, 0, 0};
  v4i g3 = {0, 0, 0, 0};
#ifdef TDM_6ARG
  v8i gz = {0, 0, 0, 0, 0, 0, 0, 0};
  __builtin_amdgcn_tensor_load_to_lds(g0, g1, g2, g3, gz, 0);
#else
  __builtin_amdgcn_tensor_load_to_lds(g0, g1, g2, g3, 0);
#endif
}

// ---------------------------------------------------------------------------
// conv1x1 GEMM, all-f16 data path:
//   inT  : f16, pixel-major [B][N][C]
//   outh : f16; transpose_out ? [B][N][O] : [B][O][N]
// Weights staged to LDS f16 once; input tiles (64 px x 32 ch) double-buffered
// via TDM. Fragments = contiguous b128 LDS reads. grid = (NPIX/64, O/16, B)
// ---------------------------------------------------------------------------
__global__ void k_conv1x1(const _Float16* __restrict__ inT, const float* __restrict__ w,
                          const float* __restrict__ bias, _Float16* __restrict__ outh,
                          int O, int transpose_out) {
  __shared__ _Float16 s_w[16 * CDIM];        // 8 KB weights (f16)
  __shared__ _Float16 s_in[2][64 * 32];      // 8 KB input double buffer [px][ch]

  const int tid  = threadIdx.x;
  const int lane = tid & 31;
  const int wave = tid >> 5;
  const int col  = lane & 15;
  const int half = lane >> 4;
  const int b    = blockIdx.z;
  const int M0   = blockIdx.y << 4;
  const int nb   = blockIdx.x << 6;
  const int nloc = (wave << 4) + col;

  const _Float16* inb = inT + (size_t)b * NPIX * CDIM;

  for (int i = tid; i < 16 * CDIM; i += 128)
    s_w[i] = (_Float16)w[(M0 + (i >> 8)) * CDIM + (i & 255)];

  if (wave == 0)
    tdm_load_2d(lds_off(&s_in[0][0]), inb + (size_t)nb * CDIM, 32u, 64u,
                (unsigned long long)CDIM, 1u);

  v8f acc = {0.f,0.f,0.f,0.f,0.f,0.f,0.f,0.f};

  for (int ks = 0; ks < 8; ++ks) {
    const int buf = ks & 1;
    if (wave == 0) {
      if (ks < 7) {
        tdm_load_2d(lds_off(&s_in[buf ^ 1][0]),
                    inb + (size_t)nb * CDIM + (ks + 1) * 32,
                    32u, 64u, (unsigned long long)CDIM, 1u);
        __builtin_amdgcn_s_wait_tensorcnt(1);
      } else {
        __builtin_amdgcn_s_wait_tensorcnt(0);
      }
    }
    __syncthreads();

    v16h a, bb;
#pragma unroll
    for (int e = 0; e < 16; ++e) {
      const int kk = kmap(half, e);
      a[e]  = s_w[col * CDIM + (ks << 5) + kk];     // A: lane = row o
      bb[e] = s_in[buf][nloc * 32 + kk];            // B: lane = col n
    }
    acc = __builtin_amdgcn_wmma_f32_16x16x32_f16(false, a, false, bb,
                                                 (short)0, acc, false, false);
    __syncthreads();
  }

  if (transpose_out) {
    _Float16* ob = outh + (size_t)b * NPIX * O + (size_t)(nb + nloc) * O;
#pragma unroll
    for (int i = 0; i < 8; ++i) {
      const int m = M0 + i + (half << 3);
      ob[m] = (_Float16)(acc[i] + bias[m]);
    }
  } else {
    _Float16* ob = outh + (size_t)b * O * NPIX;
#pragma unroll
    for (int i = 0; i < 8; ++i) {
      const int m = M0 + i + (half << 3);
      ob[(size_t)m * NPIX + nb + nloc] = (_Float16)(acc[i] + bias[m]);
    }
  }
}

// ---------------------------------------------------------------------------
// conv3x3 'SAME' as 9-tap implicit GEMM; input = f16 pixel-major [B][N][C]
// (per-lane contiguous b128 loads), output f32 [B][C][N].
// grid = (NPIX/64, 16, B), block 128
// ---------------------------------------------------------------------------
__global__ void k_conv3x3(const _Float16* __restrict__ inT, const float* __restrict__ w,
                          const float* __restrict__ bias, float* __restrict__ out) {
  __shared__ _Float16 s_w[16 * CDIM];

  const int tid  = threadIdx.x;
  const int lane = tid & 31;
  const int wave = tid >> 5;
  const int col  = lane & 15;
  const int half = lane >> 4;
  const int b    = blockIdx.z;
  const int M0   = blockIdx.y << 4;
  const int n    = (blockIdx.x << 6) + (wave << 4) + col;
  const int py   = n >> 6, px = n & 63;

  const _Float16* inb = inT + (size_t)b * NPIX * CDIM;
  v8f acc = {0.f,0.f,0.f,0.f,0.f,0.f,0.f,0.f};

  for (int tap = 0; tap < 9; ++tap) {
    __syncthreads();
    for (int i = tid; i < 16 * CDIM; i += 128)
      s_w[i] = (_Float16)w[((size_t)(M0 + (i >> 8)) * CDIM + (i & 255)) * 9 + tap];
    __syncthreads();

    const int dy = tap / 3 - 1, dx = tap % 3 - 1;
    const int ys = py + dy, xs = px + dx;
    const bool inbnd = (ys >= 0) & (ys < 64) & (xs >= 0) & (xs < 64);
    const size_t srow = (size_t)((ys << 6) + xs) * CDIM;

    for (int k0 = 0; k0 < CDIM; k0 += 32) {
      v16h a, bb;
#pragma unroll
      for (int e = 0; e < 16; ++e) {
        const int kk = kmap(half, e);
        a[e]  = s_w[col * CDIM + k0 + kk];
        bb[e] = inbnd ? inb[srow + k0 + kk] : (_Float16)0.f;
      }
      acc = __builtin_amdgcn_wmma_f32_16x16x32_f16(false, a, false, bb,
                                                   (short)0, acc, false, false);
    }
  }
  float* outb = out + (size_t)b * CDIM * NPIX;
#pragma unroll
  for (int i = 0; i < 8; ++i) {
    const int m = M0 + i + (half << 3);
    outb[(size_t)m * NPIX + n] = acc[i] + bias[m];
  }
}

// ---------------------------------------------------------------------------
// Fused flash attention, all-f16 tiles:
//   qT, kT : f16 [B][N][32] (key/query-major)   vh : f16 [B][C][N]
//   out    : f32 [B][C][N]
// K tile = one contiguous 2KB TDM burst; V tile = 256x32 TDM; both
// double-buffered. block = 256 (8 waves) per 16-query block. grid=(NPIX/16,B)
// ---------------------------------------------------------------------------
__global__ void k_flash(const _Float16* __restrict__ qT, const _Float16* __restrict__ kT,
                        const _Float16* __restrict__ vh, float* __restrict__ out) {
  __shared__ _Float16 s_k[2][32 * 32];     //  4 KB K tiles [key][d]
  __shared__ _Float16 s_v[2][CDIM * 32];   // 32 KB V tiles [c][key]
  __shared__ float    s_s[16 * 33];        // raw scores
  __shared__ _Float16 s_p[16 * 40];        // probabilities (16B-aligned rows)
  __shared__ float    s_m[16], s_l[16], s_c[16];

  const int tid  = threadIdx.x;
  const int lane = tid & 31;
  const int wave = tid >> 5;
  const int col  = lane & 15;
  const int half = lane >> 4;
  const int b    = blockIdx.y;
  const int n0   = blockIdx.x << 4;

  const _Float16* qb = qT + (size_t)b * NPIX * CQK;
  const _Float16* kb = kT + (size_t)b * NPIX * CQK;
  const _Float16* vb = vh + (size_t)b * CDIM * NPIX;

  if (tid < 16) { s_m[tid] = -3.0e38f; s_l[tid] = 0.f; }

  // Loop-invariant Q fragment (A: lane = query row): 2x b128 global loads
  v16h aq;
  if (wave < 2) {
#pragma unroll
    for (int e = 0; e < 16; ++e)
      aq[e] = qb[(size_t)(n0 + col) * CQK + kmap(half, e)];
  }

  if (wave == 0) {
    tdm_load_2d(lds_off(&s_k[0][0]), kb, 1024u, 1u, 1024ull, 1u);
    tdm_load_2d(lds_off(&s_v[0][0]), vb, 32u, (unsigned)CDIM,
                (unsigned long long)NPIX, 1u);
  }

  const int c0 = wave << 5;
  v8f acc0 = {0.f,0.f,0.f,0.f,0.f,0.f,0.f,0.f};
  v8f acc1 = {0.f,0.f,0.f,0.f,0.f,0.f,0.f,0.f};

  for (int it = 0; it < NPIX / 32; ++it) {
    const int buf = it & 1;
    if (wave == 0) {
      if (it < NPIX / 32 - 1) {
        const int mn = (it + 1) << 5;
        tdm_load_2d(lds_off(&s_k[buf ^ 1][0]), kb + (size_t)mn * CQK,
                    1024u, 1u, 1024ull, 1u);
        tdm_load_2d(lds_off(&s_v[buf ^ 1][0]), vb + mn,
                    32u, (unsigned)CDIM, (unsigned long long)NPIX, 1u);
        __builtin_amdgcn_s_wait_tensorcnt(2);
      } else {
        __builtin_amdgcn_s_wait_tensorcnt(0);
      }
    }
    __syncthreads();

    // ---- phase 1: score tiles S[16q x 32k] (waves 0,1) ----
    if (wave < 2) {
      v16h bk;
#pragma unroll
      for (int e = 0; e < 16; ++e)
        bk[e] = s_k[buf][((wave << 4) + col) * 32 + kmap(half, e)];
      v8f z = {0.f,0.f,0.f,0.f,0.f,0.f,0.f,0.f};
      v8f s = __builtin_amdgcn_wmma_f32_16x16x32_f16(false, aq, false, bk,
                                                     (short)0, z, false, false);
#pragma unroll
      for (int i = 0; i < 8; ++i)
        s_s[(i + (half << 3)) * 33 + (wave << 4) + col] = s[i];
    }
    __syncthreads();

    // ---- phase 2: online softmax row update (one lane per query row) ----
    if (tid < 16) {
      float mold = s_m[tid];
      float mx = mold;
#pragma unroll
      for (int j = 0; j < 32; ++j) mx = fmaxf(mx, s_s[tid * 33 + j]);
      float corr = __expf(mold - mx);
      float sum = 0.f;
#pragma unroll
      for (int j = 0; j < 32; ++j) {
        float p = __expf(s_s[tid * 33 + j] - mx);
        s_p[tid * 40 + j] = (_Float16)p;
        sum += p;
      }
      s_m[tid] = mx;
      s_l[tid] = s_l[tid] * corr + sum;
      s_c[tid] = corr;
    }
    __syncthreads();

    // ---- phase 3: rescale + P.V accumulation (all 8 waves) ----
    v16h ap, bv0, bv1;
#pragma unroll
    for (int e = 0; e < 16; ++e)
      ap[e] = s_p[col * 40 + kmap(half, e)];
#pragma unroll
    for (int e = 0; e < 16; ++e) {
      const int kk = kmap(half, e);
      bv0[e] = s_v[buf][(c0 + col) * 32 + kk];
      bv1[e] = s_v[buf][(c0 + 16 + col) * 32 + kk];
    }
#pragma unroll
    for (int i = 0; i < 8; ++i) {
      const float corr = s_c[i + (half << 3)];
      acc0[i] *= corr;
      acc1[i] *= corr;
    }
    acc0 = __builtin_amdgcn_wmma_f32_16x16x32_f16(false, ap, false, bv0,
                                                  (short)0, acc0, false, false);
    acc1 = __builtin_amdgcn_wmma_f32_16x16x32_f16(false, ap, false, bv1,
                                                  (short)0, acc1, false, false);
    __syncthreads();
  }

  float* ob = out + (size_t)b * CDIM * NPIX;
#pragma unroll
  for (int i = 0; i < 8; ++i) {
    const int row = i + (half << 3);
    const float il = 1.0f / s_l[row];
    const int nq = n0 + row;
    ob[(size_t)(c0 + col) * NPIX + nq]      = acc0[i] * il;
    ob[(size_t)(c0 + 16 + col) * NPIX + nq] = acc1[i] * il;
  }
}

// ---------------------------------------------------------------------------
// out = InstanceNorm( gamma*a + r ); optional f16 pixel-major mirror outh.
// one 256-thread block per (b, c) channel. grid = (C, B)
// ---------------------------------------------------------------------------
__global__ void k_res_inorm(const float* __restrict__ a, const float* __restrict__ r,
                            const float* __restrict__ gamma, float* __restrict__ out,
                            _Float16* __restrict__ outh) {
  __shared__ float s1[256], s2[256];
  const int tid = threadIdx.x;
  const int c = blockIdx.x, b = blockIdx.y;
  const size_t base = ((size_t)b * CDIM + c) * NPIX;
  const float g = gamma ? gamma[0] : 1.0f;

  float vals[16];
  float sum = 0.f, sq = 0.f;
#pragma unroll
  for (int t = 0; t < 16; ++t) {
    const int idx = tid + (t << 8);
    const float x = g * a[base + idx] + r[base + idx];
    vals[t] = x; sum += x; sq += x * x;
  }
  s1[tid] = sum; s2[tid] = sq;
  __syncthreads();
  for (int s = 128; s > 0; s >>= 1) {
    if (tid < s) { s1[tid] += s1[tid + s]; s2[tid] += s2[tid + s]; }
    __syncthreads();
  }
  const float mean = s1[0] * (1.0f / NPIX);
  const float var  = s2[0] * (1.0f / NPIX) - mean * mean;
  const float rs   = rsqrtf(var + EPSV);
#pragma unroll
  for (int t = 0; t < 16; ++t) {
    const int idx = tid + (t << 8);
    const float o = (vals[t] - mean) * rs;
    out[base + idx] = o;
    if (outh) outh[((size_t)b * NPIX + idx) * CDIM + c] = (_Float16)o;
  }
}

// ---------------------------------------------------------------------------
// f32 [B][C][N] -> f16 pixel-major [B][N][C]. grid = (NPIX/256, C, B)
// ---------------------------------------------------------------------------
__global__ void k_tohalf_t(const float* __restrict__ in, _Float16* __restrict__ out) {
  const int n = (blockIdx.x << 8) + threadIdx.x;
  const int c = blockIdx.y, b = blockIdx.z;
  out[((size_t)b * NPIX + n) * CDIM + c] = (_Float16)in[((size_t)b * CDIM + c) * NPIX + n];
}

// ---------------------------------------------------------------------------
extern "C" void kernel_launch(void* const* d_in, const int* in_sizes, int n_in,
                              void* d_out, int out_size, void* d_ws, size_t ws_size,
                              hipStream_t stream) {
  (void)in_sizes; (void)n_in; (void)out_size; (void)ws_size;

  const float* x = (const float*)d_in[0];
  const float* y = (const float*)d_in[1];
  const float* sa1_wq = (const float*)d_in[2];  const float* sa1_bq = (const float*)d_in[3];
  const float* sa1_wk = (const float*)d_in[4];  const float* sa1_bk = (const float*)d_in[5];
  const float* sa1_wv = (const float*)d_in[6];  const float* sa1_bv = (const float*)d_in[7];
  const float* sa1_g  = (const float*)d_in[8];
  const float* sa2_wq = (const float*)d_in[9];  const float* sa2_bq = (const float*)d_in[10];
  const float* sa2_wk = (const float*)d_in[11]; const float* sa2_bk = (const float*)d_in[12];
  const float* sa2_wv = (const float*)d_in[13]; const float* sa2_bv = (const float*)d_in[14];
  const float* sa2_g  = (const float*)d_in[15];
  const float* top_wq = (const float*)d_in[16]; const float* top_bq = (const float*)d_in[17];
  const float* top_wk = (const float*)d_in[18]; const float* top_bk = (const float*)d_in[19];
  const float* top_wv = (const float*)d_in[20]; const float* top_bv = (const float*)d_in[21];
  const float* top_g  = (const float*)d_in[22];
  const float* c1w = (const float*)d_in[23]; const float* c1b = (const float*)d_in[24];
  const float* c2w = (const float*)d_in[25]; const float* c2b = (const float*)d_in[26];

  const size_t CN  = (size_t)BDIM * CDIM * NPIX;   // f32 tensor elems
  const size_t QKh = (size_t)BDIM * CQK * NPIX;    // f16 q/k elems

  float* f32b = (float*)d_ws;
  float* t0 = f32b;      float* t1 = t0 + CN;  float* t2 = t1 + CN;
  float* t3 = t2 + CN;   float* t4 = t3 + CN;  float* t5 = t4 + CN;
  _Float16* hb  = (_Float16*)(t5 + CN);
  _Float16* xh  = hb;            _Float16* yh  = xh + CN;
  _Float16* vhb = yh + CN;
  _Float16* t1h = vhb + CN;      _Float16* t2h = t1h + CN;
  _Float16* t3h = t2h + CN;      _Float16* t4h = t3h + CN;
  _Float16* qh  = t4h + CN;      _Float16* kh  = qh + QKh;

  const dim3 blk128(128), blk256(256);
  const dim3 gC1_32(NPIX / 64, CQK / 16, BDIM);
  const dim3 gC1_256(NPIX / 64, CDIM / 16, BDIM);
  const dim3 gC3(NPIX / 64, CDIM / 16, BDIM);
  const dim3 gFl(NPIX / 16, BDIM);
  const dim3 gIn(CDIM, BDIM);
  const dim3 gTh(NPIX / 256, CDIM, BDIM);
  float* outp = (float*)d_out;

  // f16 pixel-major mirrors of the two external inputs
  k_tohalf_t<<<gTh, blk256, 0, stream>>>(x, xh);
  k_tohalf_t<<<gTh, blk256, 0, stream>>>(y, yh);

  // ---- stage 1: out_1 = inorm(self_attn(x, sa1)) -> t1 (+t1h) ----
  k_conv1x1<<<gC1_32,  blk128, 0, stream>>>(xh, sa1_wq, sa1_bq, qh,  CQK, 1);
  k_conv1x1<<<gC1_32,  blk128, 0, stream>>>(xh, sa1_wk, sa1_bk, kh,  CQK, 1);
  k_conv1x1<<<gC1_256, blk128, 0, stream>>>(xh, sa1_wv, sa1_bv, vhb, CDIM, 0);
  k_flash  <<<gFl,     blk256, 0, stream>>>(qh, kh, vhb, t0);
  k_res_inorm<<<gIn,   blk256, 0, stream>>>(t0, x, sa1_g, t1, t1h);

  // ---- stage 2: out_2 = inorm(out_1 + conv3x3(out_1)) -> t2 (+t2h) ----
  k_conv3x3<<<gC3,     blk128, 0, stream>>>(t1h, c1w, c1b, t4);
  k_res_inorm<<<gIn,   blk256, 0, stream>>>(t4, t1, nullptr, t2, t2h);

  // ---- stage 3: out_3 = inorm(self_attn(y, sa2)) -> t3 (+t3h) ----
  k_conv1x1<<<gC1_32,  blk128, 0, stream>>>(yh, sa2_wq, sa2_bq, qh,  CQK, 1);
  k_conv1x1<<<gC1_32,  blk128, 0, stream>>>(yh, sa2_wk, sa2_bk, kh,  CQK, 1);
  k_conv1x1<<<gC1_256, blk128, 0, stream>>>(yh, sa2_wv, sa2_bv, vhb, CDIM, 0);
  k_flash  <<<gFl,     blk256, 0, stream>>>(qh, kh, vhb, t0);
  k_res_inorm<<<gIn,   blk256, 0, stream>>>(t0, y, sa2_g, t3, t3h);

  // ---- stage 4: top cross-attention (q,k from out_2; v from out_3) ----
  k_conv1x1<<<gC1_32,  blk128, 0, stream>>>(t2h, top_wq, top_bq, qh,  CQK, 1);
  k_conv1x1<<<gC1_32,  blk128, 0, stream>>>(t2h, top_wk, top_bk, kh,  CQK, 1);
  k_conv1x1<<<gC1_256, blk128, 0, stream>>>(t3h, top_wv, top_bv, vhb, CDIM, 0);
  k_flash  <<<gFl,     blk256, 0, stream>>>(qh, kh, vhb, t0);
  k_res_inorm<<<gIn,   blk256, 0, stream>>>(t0, y, top_g, t5, nullptr);   // inorm(g*att + y)
  k_res_inorm<<<gIn,   blk256, 0, stream>>>(t5, t3, nullptr, t4, t4h);    // inorm(. + out_3)
  k_conv3x3<<<gC3,     blk128, 0, stream>>>(t4h, c2w, c2b, t0);           // out_4
  k_res_inorm<<<gIn,   blk256, 0, stream>>>(t0, t4, nullptr, outp, nullptr); // final
}
